// Phi4FlashDecoderLayer_56650618635027
// MI455X (gfx1250) — matrix operand order repaired and checked by reading, not compiled, not run on hardware
//
#include <hip/hip_runtime.h>
#include <hip/hip_bf16.h>
#include <stdint.h>

// ---------------------------------------------------------------------------
// MI455X (gfx1250) Phi4 differential-attention layer.
//  - bf16 WMMA (v_wmma_f32_16x16x32_bf16) for QKV GEMM, attention, out GEMM
//  - GEMMs: B tile shared through TRIPLE-buffered LDS filled with
//    global_load_async_to_lds_b128 (ASYNCcnt pipeline, one barrier per
//    32-k step); k-loop unrolled 3x with rotating A fragments (no moves).
//  - attention: fused flash kernel, never materializes 20x2048x2048 probs.
// ---------------------------------------------------------------------------

typedef __attribute__((ext_vector_type(16))) __bf16 v16bf;
typedef __attribute__((ext_vector_type(8)))  float  v8f;
typedef __attribute__((ext_vector_type(4)))  int    v4i;
typedef __attribute__((address_space(1))) v4i g_v4i;   // global 128b chunk
typedef __attribute__((address_space(3))) v4i l_v4i;   // LDS 128b chunk

#define S_LEN 2048
#define HID   2560
#define HD    64
#define QKV_N 3072          // 40*64 + 2*4*64
#define NPAIR 20
#define WIN   2047

__device__ __forceinline__ unsigned short f2bf(float f) {
    unsigned int u = __float_as_uint(f);
    unsigned int r = u + 0x7FFFu + ((u >> 16) & 1u);   // round-to-nearest-even
    return (unsigned short)(r >> 16);
}

__device__ __forceinline__ v8f zero8() {
    v8f z;
#pragma unroll
    for (int i = 0; i < 8; ++i) z[i] = 0.0f;
    return z;
}

__device__ __forceinline__ v8f wmma_bf16(v16bf a, v16bf b, v8f c) {
    return __builtin_amdgcn_wmma_f32_16x16x32_bf16(false, a, false, b,
                                                   (short)0, c, false, false);
}

union FragU {
    v16bf v;
    uint4 q[2];
};

// ---- CDNA5 async global->LDS copy (ASYNCcnt) ------------------------------
__device__ __forceinline__ void async_copy_b128(unsigned short* lds_dst,
                                                const unsigned short* g_src) {
#if __has_builtin(__builtin_amdgcn_global_load_async_to_lds_b128)
    __builtin_amdgcn_global_load_async_to_lds_b128(
        (g_v4i*)(uintptr_t)g_src,
        (l_v4i*)(uint32_t)(uintptr_t)lds_dst, 0, 0);
#else
    unsigned lds_off = (unsigned)(uintptr_t)lds_dst;
    asm volatile("global_load_async_to_lds_b128 %0, %1, off"
                 :: "v"(lds_off), "v"(g_src) : "memory");
#endif
}

__device__ __forceinline__ void wait_async_le1() {
#if __has_builtin(__builtin_amdgcn_s_wait_asynccnt)
    __builtin_amdgcn_s_wait_asynccnt(1);
#else
    asm volatile("s_wait_asynccnt 0x1" ::: "memory");
#endif
}
__device__ __forceinline__ void wait_async_le0() {
#if __has_builtin(__builtin_amdgcn_s_wait_asynccnt)
    __builtin_amdgcn_s_wait_asynccnt(0);
#else
    asm volatile("s_wait_asynccnt 0x0" ::: "memory");
#endif
}

// ---------------------------------------------------------------------------
// fp32 -> bf16 elementwise conversion
// ---------------------------------------------------------------------------
__global__ void k_f32_to_bf16(const float* __restrict__ src,
                              unsigned short* __restrict__ dst, int n) {
    int i = blockIdx.x * blockDim.x + threadIdx.x;
    if (i < n) dst[i] = f2bf(src[i]);
}

// ---------------------------------------------------------------------------
// C[M,N] = A[M,K](bf16) * W[N,K](bf16)^T + bias[N]   (fp32 out)
// Block: 256 threads = 8 waves; block tile 128x64 (wave strip 16x64).
// B tile (64 cols x 32 k) triple-buffered in LDS via async DMA, shared by
// all 8 waves. One barrier per 32-k step: after barrier(s) every wave has
// finished reading buffer (s-1)%3, so it can be refilled for step s+2.
// Requires K/32 >= 3 and K/32 == 2 (mod 3)  (K = 2560 -> 80 steps).
// LDS B row stride = 40 bf16 (80B): bank-spread + 16B aligned chunks.
// ---------------------------------------------------------------------------
__device__ __forceinline__ void read_bfrags(FragU bf[4],
                                            const unsigned short* buf,
                                            int half, int l16) {
    const unsigned short* bbase = buf + (half ? 8 : 0);
#pragma unroll
    for (int t = 0; t < 4; ++t) {
        const unsigned short* bp = bbase + (t * 16 + l16) * 40;
        bf[t].q[0] = *(const uint4*)(bp);
        bf[t].q[1] = *(const uint4*)(bp + 16);
    }
}

__global__ __launch_bounds__(256) void k_gemm_bf16(
    const unsigned short* __restrict__ A,   // [M,K]
    const unsigned short* __restrict__ W,   // [N,K]
    const float* __restrict__ bias,         // [N]
    float* __restrict__ C,                  // [M,N]
    int M, int N, int K)
{
    __shared__ unsigned short Bt[3][64 * 40];

    const int lane = threadIdx.x & 31;
    const int wave = threadIdx.x >> 5;
    const int half = lane >> 4;
    const int l16  = lane & 15;
    const int mbase = blockIdx.y * 128 + wave * 16;
    const int nbase = blockIdx.x * 64;

    // per-thread slice of the B-tile DMA: 64 rows x 4 chunks of 8 bf16 (16B)
    const int frow = threadIdx.x >> 2;
    const int fch  = (threadIdx.x & 3) * 8;
    const unsigned short* fgp = W + (size_t)(nbase + frow) * K + fch;
    unsigned short* flp[3];
#pragma unroll
    for (int j = 0; j < 3; ++j) flp[j] = &Bt[j][0] + frow * 40 + fch;

    const unsigned short* aptr = A + (size_t)(mbase + l16) * K + (half ? 8 : 0);

    v8f acc[4];
#pragma unroll
    for (int t = 0; t < 4; ++t) acc[t] = zero8();

    const int nsteps = K >> 5;                 // 32-k steps (80 for K=2560)

    // preamble: DMA steps 0 and 1 (step 2's fill is issued inside step 0),
    // load A fragment for step 0
    async_copy_b128(flp[0], fgp);
    async_copy_b128(flp[1], fgp + 32);
    FragU aF[3];
    aF[0].q[0] = *(const uint4*)(aptr + 0);
    aF[0].q[1] = *(const uint4*)(aptr + 16);

    // one 32-k step at buffer j (== s % 3): wait fill(s), barrier, read
    // frags, refill buf (j+2)%3 for step s+2, prefetch A for step s+1,
    // then 4 WMMAs.  j must be a literal so aF indexing constant-folds.
    auto step = [&](int j, int s, bool doFill, bool doALoad, bool last) {
        if (last) wait_async_le0(); else wait_async_le1();
        __syncthreads();
        FragU bf[4];
        read_bfrags(bf, &Bt[j][0], half, l16);
        if (doFill)
            async_copy_b128(flp[(j + 2) % 3], fgp + (size_t)(s + 2) * 32);
        if (doALoad) {
            const int jn = (j + 1) % 3;
            aF[jn].q[0] = *(const uint4*)(aptr + (size_t)(s + 1) * 32);
            aF[jn].q[1] = *(const uint4*)(aptr + (size_t)(s + 1) * 32 + 16);
        }
#pragma unroll
        for (int t = 0; t < 4; ++t)
            acc[t] = wmma_bf16(aF[j].v, bf[t].v, acc[t]);
    };

    // steady state: unrolled by 3, fully branch-free
    for (int s = 0; s < nsteps - 2; s += 3) {
        step(0, s,     true, true, false);
        step(1, s + 1, true, true, false);
        step(2, s + 2, true, true, false);
    }
    // tail: last two steps, no refills  (valid since nsteps % 3 == 2)
    step(0, nsteps - 2, false, true,  false);
    step(1, nsteps - 1, false, false, true);

#pragma unroll
    for (int t = 0; t < 4; ++t) {
        const int n  = nbase + t * 16 + l16;
        const float bv = bias[n];
#pragma unroll
        for (int r = 0; r < 8; ++r) {
            const int m = mbase + r + (half ? 8 : 0);
            C[(size_t)m * N + n] = acc[t][r] + bv;
        }
    }
}

// ---------------------------------------------------------------------------
// Split qkv fp32 [S,3072] into bf16 Q [S,2560], K [S,256], Vt [256,S]
// ---------------------------------------------------------------------------
__global__ void k_postproc(const float* __restrict__ qkv,
                           unsigned short* __restrict__ Qb,
                           unsigned short* __restrict__ Kb,
                           unsigned short* __restrict__ Vt)
{
    int idx = blockIdx.x * blockDim.x + threadIdx.x;
    if (idx >= S_LEN * QKV_N) return;
    int s = idx / QKV_N, c = idx % QKV_N;
    unsigned short b = f2bf(qkv[idx]);
    if (c < HID)            Qb[(size_t)s * HID + c] = b;
    else if (c < HID + 256) Kb[(size_t)s * 256 + (c - HID)] = b;
    else                    Vt[(size_t)(c - HID - 256) * S_LEN + s] = b;
}

// ---------------------------------------------------------------------------
// Fused differential flash attention.
// Grid: (128 query tiles, 20 paired heads). Block: 64 threads = 2 waves
// (wave = softmax set). Per 32-key step: issue K frags, then v1 frags, score
// WMMAs (partial loadcnt wait leaves V in flight through the softmax), LDS
// restage of P (C-layout -> A-layout), issue v2 frags, 8 PV WMMAs.
// ---------------------------------------------------------------------------
__global__ __launch_bounds__(64) void k_diffattn(
    const unsigned short* __restrict__ Qb,   // [S,2560]
    const unsigned short* __restrict__ Kb,   // [S,256]
    const unsigned short* __restrict__ Vt,   // [256,S]
    const float* __restrict__ lq1, const float* __restrict__ lk1,
    const float* __restrict__ lq2, const float* __restrict__ lk2,
    const float* __restrict__ sublnw,        // [128]
    unsigned short* __restrict__ attnOut)    // [S,2560] bf16
{
    __shared__ unsigned short ldsP[2][16 * 40];
    __shared__ float ldsO[2][16][128];

    const int qt   = blockIdx.x;
    const int h    = blockIdx.y;
    const int wave = threadIdx.x >> 5;
    const int lane = threadIdx.x & 31;
    const int half = lane >> 4;
    const int l16  = lane & 15;
    const int q0   = qt * 16;
    const int kvp  = h / 10;

    const int qcol = (2 * h   + wave) * HD;
    const int kcol = (2 * kvp + wave) * HD;
    const int vrow[2] = { (2 * kvp) * HD, (2 * kvp + 1) * HD };

    const unsigned short* qp =
        Qb + (size_t)(q0 + l16) * HID + qcol + (half ? 8 : 0);
    FragU qf[2];
    qf[0].q[0] = *(const uint4*)(qp + 0);
    qf[0].q[1] = *(const uint4*)(qp + 16);
    qf[1].q[0] = *(const uint4*)(qp + 32);
    qf[1].q[1] = *(const uint4*)(qp + 48);

    float mrow[8], lrow[8];
#pragma unroll
    for (int r = 0; r < 8; ++r) { mrow[r] = -3.0e38f; lrow[r] = 0.0f; }
    v8f Oa[2][4];
#pragma unroll
    for (int v = 0; v < 2; ++v)
#pragma unroll
        for (int dt = 0; dt < 4; ++dt) Oa[v][dt] = zero8();

    unsigned short* myP = ldsP[wave];
    const int kmax = (qt + 1) * 16;

    for (int kb = 0; kb < kmax; kb += 32) {
        // ---- issue K fragments (needed first) ----
        FragU kf[2][2];
#pragma unroll
        for (int ct = 0; ct < 2; ++ct) {
            const unsigned short* kp =
                Kb + (size_t)(kb + ct * 16 + l16) * 256 + kcol + (half ? 8 : 0);
            kf[ct][0].q[0] = *(const uint4*)(kp);
            kf[ct][0].q[1] = *(const uint4*)(kp + 16);
            kf[ct][1].q[0] = *(const uint4*)(kp + 32);
            kf[ct][1].q[1] = *(const uint4*)(kp + 48);
        }
        // ---- issue v1 fragments early (consumed after softmax) ----
        FragU vf0[4];
#pragma unroll
        for (int dt = 0; dt < 4; ++dt) {
            const unsigned short* vp =
                Vt + (size_t)(vrow[0] + dt * 16 + l16) * S_LEN + kb +
                (half ? 8 : 0);
            vf0[dt].q[0] = *(const uint4*)(vp);
            vf0[dt].q[1] = *(const uint4*)(vp + 16);
        }
        // ---- scores ----
        v8f sc[2];
#pragma unroll
        for (int ct = 0; ct < 2; ++ct) {
            v8f s = zero8();
            s = wmma_bf16(qf[0].v, kf[ct][0].v, s);
            s = wmma_bf16(qf[1].v, kf[ct][1].v, s);
            const int j = kb + ct * 16 + l16;
#pragma unroll
            for (int r = 0; r < 8; ++r) {
                const int i = q0 + r + (half ? 8 : 0);
                const bool ok = (j <= i) && ((i - j) < WIN);
                s[r] = ok ? s[r] * 0.125f : -3.0e38f;
            }
            sc[ct] = s;
        }
        // ---- online softmax (row stats per VGPR slot, 16-lane shfl) ----
        float fac[8];
#pragma unroll
        for (int r = 0; r < 8; ++r) {
            float t = fmaxf(sc[0][r], sc[1][r]);
#pragma unroll
            for (int m = 1; m < 16; m <<= 1) t = fmaxf(t, __shfl_xor(t, m));
            const float mnew = fmaxf(mrow[r], t);
            const float f    = __expf(mrow[r] - mnew);
            const float p0   = __expf(sc[0][r] - mnew);
            const float p1   = __expf(sc[1][r] - mnew);
            sc[0][r] = p0; sc[1][r] = p1;
            float rs = p0 + p1;
#pragma unroll
            for (int m = 1; m < 16; m <<= 1) rs += __shfl_xor(rs, m);
            lrow[r] = lrow[r] * f + rs;
            mrow[r] = mnew; fac[r] = f;
        }
#pragma unroll
        for (int v = 0; v < 2; ++v)
#pragma unroll
            for (int dt = 0; dt < 4; ++dt)
#pragma unroll
                for (int r = 0; r < 8; ++r) Oa[v][dt][r] *= fac[r];

        // ---- restage P through wave-private LDS: C-layout -> A-layout ----
#pragma unroll
        for (int ct = 0; ct < 2; ++ct)
#pragma unroll
            for (int r = 0; r < 8; ++r) {
                const int m = r + (half ? 8 : 0);
                myP[m * 40 + ct * 16 + l16] = f2bf(sc[ct][r]);
            }
        FragU pa;
        const unsigned short* pp = myP + (size_t)l16 * 40 + (half ? 8 : 0);
        pa.q[0] = *(const uint4*)(pp);
        pa.q[1] = *(const uint4*)(pp + 16);

        // ---- issue v2 fragments; they complete under the v1 WMMAs ----
        FragU vf1[4];
#pragma unroll
        for (int dt = 0; dt < 4; ++dt) {
            const unsigned short* vp =
                Vt + (size_t)(vrow[1] + dt * 16 + l16) * S_LEN + kb +
                (half ? 8 : 0);
            vf1[dt].q[0] = *(const uint4*)(vp);
            vf1[dt].q[1] = *(const uint4*)(vp + 16);
        }
#pragma unroll
        for (int dt = 0; dt < 4; ++dt)
            Oa[0][dt] = wmma_bf16(pa.v, vf0[dt].v, Oa[0][dt]);
#pragma unroll
        for (int dt = 0; dt < 4; ++dt)
            Oa[1][dt] = wmma_bf16(pa.v, vf1[dt].v, Oa[1][dt]);
    }

    // ---- normalize & publish this set's 16x128 output ----
#pragma unroll
    for (int v = 0; v < 2; ++v)
#pragma unroll
        for (int dt = 0; dt < 4; ++dt)
#pragma unroll
            for (int r = 0; r < 8; ++r) {
                const int m = r + (half ? 8 : 0);
                ldsO[wave][m][v * 64 + dt * 16 + l16] = Oa[v][dt][r] / lrow[r];
            }
    __syncthreads();

    // ---- differential combine + RMSNorm(128) + scale, bf16 store ----
    float lam1 = 0.f, lam2 = 0.f;
    for (int i = 0; i < 64; ++i) {
        lam1 += lq1[i] * lk1[i];
        lam2 += lq2[i] * lk2[i];
    }
    const float LAMBDA_INIT = 0.8f - 0.6f * __expf(-0.3f);
    const float lamf = __expf(lam1) - __expf(lam2) + LAMBDA_INIT;

    const int row   = threadIdx.x >> 2;
    const int cpart = threadIdx.x & 3;
    float vals[32];
    float ss = 0.f;
#pragma unroll
    for (int c = 0; c < 32; ++c) {
        const int col = cpart * 32 + c;
        const float d = ldsO[0][row][col] - lamf * ldsO[1][row][col];
        vals[c] = d; ss += d * d;
    }
    ss += __shfl_xor(ss, 1);
    ss += __shfl_xor(ss, 2);
    const float nrm = rsqrtf(ss * (1.0f / 128.0f) + 1e-5f) * (1.0f - LAMBDA_INIT);

    unsigned short* op = attnOut + (size_t)(q0 + row) * HID + h * 128;
#pragma unroll
    for (int c = 0; c < 32; ++c) {
        const int col = cpart * 32 + c;
        op[col] = f2bf(vals[c] * nrm * sublnw[col]);
    }
}

// ---------------------------------------------------------------------------
extern "C" void kernel_launch(void* const* d_in, const int* in_sizes, int n_in,
                              void* d_out, int out_size, void* d_ws, size_t ws_size,
                              hipStream_t stream) {
    (void)in_sizes; (void)n_in; (void)out_size; (void)ws_size;
    const float* hidden = (const float*)d_in[0];
    const float* Wqkv_w = (const float*)d_in[1];
    const float* Wqkv_b = (const float*)d_in[2];
    const float* out_w  = (const float*)d_in[3];
    const float* out_b  = (const float*)d_in[4];
    const float* lq1    = (const float*)d_in[5];
    const float* lk1    = (const float*)d_in[6];
    const float* lq2    = (const float*)d_in[7];
    const float* lk2    = (const float*)d_in[8];
    const float* sublnw = (const float*)d_in[9];
    float* out = (float*)d_out;

    char* ws = (char*)d_ws;
    size_t off = 0;
    auto alloc = [&](size_t bytes) -> void* {
        void* p = ws + off;
        off = (off + bytes + 255) & ~(size_t)255;
        return p;
    };
    unsigned short* Ab    = (unsigned short*)alloc((size_t)S_LEN * HID * 2);
    unsigned short* Wqkvb = (unsigned short*)alloc((size_t)QKV_N * HID * 2);
    unsigned short* Woutb = (unsigned short*)alloc((size_t)HID * HID * 2);
    float*          qkvf  = (float*)alloc((size_t)S_LEN * QKV_N * 4);
    unsigned short* Qb    = (unsigned short*)alloc((size_t)S_LEN * HID * 2);
    unsigned short* Kb    = (unsigned short*)alloc((size_t)S_LEN * 256 * 2);
    unsigned short* Vt    = (unsigned short*)alloc((size_t)256 * S_LEN * 2);
    unsigned short* attnb = (unsigned short*)alloc((size_t)S_LEN * HID * 2);

    {
        int n = S_LEN * HID;
        k_f32_to_bf16<<<(n + 255) / 256, 256, 0, stream>>>(hidden, Ab, n);
        n = QKV_N * HID;
        k_f32_to_bf16<<<(n + 255) / 256, 256, 0, stream>>>(Wqkv_w, Wqkvb, n);
        n = HID * HID;
        k_f32_to_bf16<<<(n + 255) / 256, 256, 0, stream>>>(out_w, Woutb, n);
    }
    // QKV projection: [2048,3072] = Ab[2048,2560] * Wqkv^T + b
    k_gemm_bf16<<<dim3(QKV_N / 64, S_LEN / 128), 256, 0, stream>>>(
        Ab, Wqkvb, Wqkv_b, qkvf, S_LEN, QKV_N, HID);
    {
        int n = S_LEN * QKV_N;
        k_postproc<<<(n + 255) / 256, 256, 0, stream>>>(qkvf, Qb, Kb, Vt);
    }
    k_diffattn<<<dim3(S_LEN / 16, NPAIR), 64, 0, stream>>>(
        Qb, Kb, Vt, lq1, lk1, lq2, lk2, sublnw, attnb);
    // output projection: out[2048,2560] = attnb * out_w^T + out_b
    k_gemm_bf16<<<dim3(HID / 64, S_LEN / 128), 256, 0, stream>>>(
        attnb, Woutb, out_b, out, S_LEN, HID, HID);
}
